// RecursiveMambaBlock_82411832475836
// MI455X (gfx1250) — matrix operand order repaired and checked by reading, not compiled, not run on hardware
//
#include <hip/hip_runtime.h>
#include <math.h>

#define L_SEQ   2048
#define DM      512
#define DI      1024
#define DS      16
#define DTR     32
#define NHEADS  8
#define HD      64

typedef __attribute__((ext_vector_type(16))) __bf16 v16bf;
typedef __attribute__((ext_vector_type(8)))  float  v8f;

union Frag { v16bf v; unsigned int u[8]; unsigned short h[16]; };

__device__ __forceinline__ unsigned short f2bf(float f) {
    unsigned u = __float_as_uint(f);
    u += 0x7fffu + ((u >> 16) & 1u);          // round-to-nearest-even
    return (unsigned short)(u >> 16);
}
__device__ __forceinline__ unsigned pack2(float lo, float hi) {
    return (unsigned)f2bf(lo) | ((unsigned)f2bf(hi) << 16);
}
__device__ __forceinline__ float sigmoidf_(float x){ return 1.f/(1.f+__expf(-x)); }
__device__ __forceinline__ float siluf(float x){ return x*sigmoidf_(x); }
__device__ __forceinline__ float geluf(float x){ return 0.5f*x*(1.f+erff(x*0.7071067811865475f)); }
__device__ __forceinline__ float softplusf(float x){ return x>20.f ? x : log1pf(__expf(x)); }
__device__ __forceinline__ float actf(float x,int a){
    if(a==1) return geluf(x);
    if(a==2) return siluf(x);
    if(a==3) return softplusf(x);
    return x;
}

// ---------------------------------------------------------------------------
// Generic bf16 WMMA GEMM: C[M,N] = act(A[M,K] @ B[K,N] + bias), fp32 in/out.
// 128 threads = 4 waves; tile 64x64, K-step 32. fp32->bf16 during LDS staging.
// Staging: global_load_b128 via uniform-base + 32-bit voffset (no flat path),
// offsets bumped by a constant per K-step.
// Requires M%64==0, N%64==0, K%32==0, lda/ldb%4==0 (true for this model).
// ---------------------------------------------------------------------------
#define BM 64
#define BN 64
#define BK 32
#define GT 128
#define LSTRIDE 34   // ushort stride (68B: dword-aligned, odd dword count)

__global__ __launch_bounds__(GT)
void k_gemm(const float* __restrict__ A, int lda,
            const float* __restrict__ B, int ldb,
            float* __restrict__ C, int ldc,
            const float* __restrict__ bias,
            int M, int N, int K, int act)
{
    __shared__ unsigned short Al[BM * LSTRIDE];
    __shared__ unsigned short Bl[BN * LSTRIDE];   // stored transposed: [n][k]

    const int tid  = threadIdx.x;
    const int lane = tid & 31;
    const int wave = tid >> 5;
    const int hlf  = lane >> 4;
    const int l16  = lane & 15;
    const int m0   = blockIdx.y * BM;
    const int n0   = blockIdx.x * BN;

    // per-thread staging coordinates (4 quads each for A and B tiles),
    // addressed as uniform base + 32-bit element offset -> global_load saddr form
    int ar[4], ac4[4], br[4], bc4[4];
    unsigned aoff[4], boff[4];
    #pragma unroll
    for (int i = 0; i < 4; ++i) {
        int ia = tid + i * GT;                    // 0..511 quads of A (64x32)
        ar[i]  = ia >> 3;                         // row 0..63
        ac4[i] = (ia & 7) * 4;                    // k 0..28
        aoff[i] = (unsigned)((m0 + ar[i]) * lda + ac4[i]);
        int ib = tid + i * GT;                    // 0..511 quads of B (32x64)
        br[i]  = ib >> 4;                         // k 0..31
        bc4[i] = (ib & 15) * 4;                   // n 0..60
        boff[i] = (unsigned)(br[i] * ldb + n0 + bc4[i]);
    }
    const unsigned bstep = (unsigned)(BK * ldb);

    v8f acc[4];
    for (int j = 0; j < 4; ++j)
        for (int e = 0; e < 8; ++e) acc[j][e] = 0.f;

    for (int kb = 0; kb < K; kb += BK) {
        __syncthreads();
        #pragma unroll
        for (int i = 0; i < 4; ++i) {
            float4 a4 = *(const float4*)(A + aoff[i]);
            aoff[i] += BK;
            unsigned* dst = (unsigned*)&Al[ar[i] * LSTRIDE + ac4[i]];
            dst[0] = pack2(a4.x, a4.y);
            dst[1] = pack2(a4.z, a4.w);
        }
        #pragma unroll
        for (int i = 0; i < 4; ++i) {
            float4 b4 = *(const float4*)(B + boff[i]);
            boff[i] += bstep;
            Bl[(bc4[i] + 0) * LSTRIDE + br[i]] = f2bf(b4.x);
            Bl[(bc4[i] + 1) * LSTRIDE + br[i]] = f2bf(b4.y);
            Bl[(bc4[i] + 2) * LSTRIDE + br[i]] = f2bf(b4.z);
            Bl[(bc4[i] + 3) * LSTRIDE + br[i]] = f2bf(b4.w);
        }
        if (kb + BK < K) {                         // global_prefetch_b8 of next A tile
            __builtin_prefetch(A + aoff[0], 0, 1);
        }
        __syncthreads();

        Frag fa;
        const int arow = wave * 16 + l16;
        #pragma unroll
        for (int j = 0; j < 8; ++j) {
            int k0 = 16 * (j >> 2) + (hlf ? 8 : 0) + 2 * (j & 3);
            fa.u[j] = *(const unsigned*)&Al[arow * LSTRIDE + k0];
        }
        #pragma unroll
        for (int j = 0; j < 4; ++j) {
            Frag fb;
            const int brow = j * 16 + l16;
            #pragma unroll
            for (int q = 0; q < 8; ++q) {
                int c0 = 16 * (q >> 2) + (hlf ? 8 : 0) + 2 * (q & 3);
                fb.u[q] = *(const unsigned*)&Bl[brow * LSTRIDE + c0];
            }
            acc[j] = __builtin_amdgcn_wmma_f32_16x16x32_bf16(
                false, fa.v, false, fb.v, (short)0, acc[j], false, false);
        }
    }

    #pragma unroll
    for (int j = 0; j < 4; ++j) {
        const int n  = n0 + j * 16 + l16;
        const float bv = bias ? bias[n] : 0.f;
        #pragma unroll
        for (int v = 0; v < 8; ++v) {
            const int mr = m0 + wave * 16 + v + 8 * hlf;
            C[(size_t)mr * ldc + n] = actf(acc[j][v] + bv, act);
        }
    }
}

// ---------------------------------------------------------------------------
// WMMA flash-attention: one wave per (16-query tile, head). Online softmax.
// qkv: [Lp, 1536] (q|k|v).  out: [Lp, 512].
// ---------------------------------------------------------------------------
__global__ __launch_bounds__(32)
void k_attn(const float* __restrict__ qkv, int Lp, float* __restrict__ out)
{
    __shared__ unsigned short Pl[16 * 32];        // bf16 P staging (C-layout -> A-layout)
    const int lane = threadIdx.x;
    const int hlf  = lane >> 4;
    const int l16  = lane & 15;
    const int q0   = blockIdx.x * 16;
    const int head = blockIdx.y;
    const int ld   = 3 * DM;
    const float scale = 0.125f;                   // 1/sqrt(64)

    Frag qa[2];
    #pragma unroll
    for (int f = 0; f < 2; ++f)
        #pragma unroll
        for (int j = 0; j < 8; ++j) {
            int k0 = 32 * f + 16 * (j >> 2) + (hlf ? 8 : 0) + 2 * (j & 3);
            unsigned qoff = (unsigned)((q0 + l16) * ld + head * HD + k0);
            float2 t = *(const float2*)(qkv + qoff);
            qa[f].u[j] = pack2(t.x, t.y);
        }

    v8f accO[4];
    for (int t = 0; t < 4; ++t) for (int e = 0; e < 8; ++e) accO[t][e] = 0.f;
    float rmax[8], rsum[8];
    for (int v = 0; v < 8; ++v) { rmax[v] = -1e30f; rsum[v] = 0.f; }

    for (int k0 = 0; k0 < Lp; k0 += 32) {
        v8f sc[2];
        #pragma unroll
        for (int sub = 0; sub < 2; ++sub) {
            for (int e = 0; e < 8; ++e) sc[sub][e] = 0.f;
            const int kb = k0 + sub * 16;
            #pragma unroll
            for (int f = 0; f < 2; ++f) {
                Frag kf;
                #pragma unroll
                for (int q = 0; q < 8; ++q) {
                    int c0 = 32 * f + 16 * (q >> 2) + (hlf ? 8 : 0) + 2 * (q & 3);
                    unsigned koff = (unsigned)((kb + l16) * ld + DM + head * HD + c0);
                    float2 t = *(const float2*)(qkv + koff);
                    kf.u[q] = pack2(t.x, t.y);
                }
                sc[sub] = __builtin_amdgcn_wmma_f32_16x16x32_bf16(
                    false, qa[f].v, false, kf.v, (short)0, sc[sub], false, false);
            }
            for (int e = 0; e < 8; ++e) sc[sub][e] *= scale;
        }

        // online softmax over this 32-key chunk (row stats reduced over 16-lane halves)
        for (int v = 0; v < 8; ++v) {
            float x = fmaxf(sc[0][v], sc[1][v]);
            for (int mk = 1; mk < 16; mk <<= 1) x = fmaxf(x, __shfl_xor(x, mk, 16));
            const float nm = fmaxf(rmax[v], x);
            const float alpha = __expf(rmax[v] - nm);
            rmax[v] = nm;
            const float p0 = __expf(sc[0][v] - nm);
            const float p1 = __expf(sc[1][v] - nm);
            sc[0][v] = p0; sc[1][v] = p1;
            float rs = p0 + p1;
            for (int mk = 1; mk < 16; mk <<= 1) rs += __shfl_xor(rs, mk, 16);
            rsum[v] = rsum[v] * alpha + rs;
            for (int t = 0; t < 4; ++t) accO[t][v] = accO[t][v] * alpha;
        }

        __syncthreads();                           // prior P reads done
        #pragma unroll
        for (int sub = 0; sub < 2; ++sub)
            #pragma unroll
            for (int v = 0; v < 8; ++v) {
                int row = v + 8 * hlf;
                Pl[row * 32 + sub * 16 + l16] = f2bf(sc[sub][v]);
            }
        __syncthreads();

        Frag pa;
        #pragma unroll
        for (int j = 0; j < 8; ++j) {
            int kk = 16 * (j >> 2) + (hlf ? 8 : 0) + 2 * (j & 3);
            pa.u[j] = *(const unsigned*)&Pl[l16 * 32 + kk];
        }
        #pragma unroll
        for (int t = 0; t < 4; ++t) {
            Frag vb;
            #pragma unroll
            for (int q = 0; q < 8; ++q) {
                int c0 = 16 * (q >> 2) + (hlf ? 8 : 0) + 2 * (q & 3);
                unsigned voff = (unsigned)((k0 + c0) * ld + 2 * DM + head * HD + t * 16 + l16);
                vb.u[q] = pack2(qkv[voff], qkv[voff + ld]);
            }
            accO[t] = __builtin_amdgcn_wmma_f32_16x16x32_bf16(
                false, pa.v, false, vb.v, (short)0, accO[t], false, false);
        }
    }

    #pragma unroll
    for (int t = 0; t < 4; ++t)
        #pragma unroll
        for (int v = 0; v < 8; ++v) {
            int row = q0 + v + 8 * hlf;
            out[(size_t)row * DM + head * HD + t * 16 + l16] = accO[t][v] / rsum[v];
        }
}

// ---------------------------------------------------------------------------
// Glue kernels
// ---------------------------------------------------------------------------
__global__ void k_conv_silu(const float* __restrict__ xz, const float* __restrict__ w,
                            const float* __restrict__ b, float* __restrict__ out)
{
    int i = blockIdx.x * blockDim.x + threadIdx.x;
    if (i >= L_SEQ * DI) return;
    int t = i / DI, ch = i % DI;
    float acc = b[ch];
    #pragma unroll
    for (int k = 0; k < 4; ++k) {
        int tt = t - 3 + k;
        if (tt >= 0) acc += xz[(size_t)tt * (2 * DI) + ch] * w[ch * 4 + k];
    }
    out[i] = siluf(acc);
}

__global__ void k_scan(const float* __restrict__ delta, const float* __restrict__ u,
                       const float* __restrict__ dBC, const float* __restrict__ A_log,
                       float* __restrict__ ys)
{
    int ch = blockIdx.x * blockDim.x + threadIdx.x;   // 0..1023
    float a[DS], h[DS];
    #pragma unroll
    for (int s = 0; s < DS; ++s) { a[s] = -__expf(A_log[ch * DS + s]); h[s] = 0.f; }
    for (int t = 0; t < L_SEQ; ++t) {
        float d  = delta[(size_t)t * DI + ch];
        float du = d * u[(size_t)t * DI + ch];
        const float* bc = dBC + (size_t)t * 64;
        float y = 0.f;
        #pragma unroll
        for (int s = 0; s < DS; ++s) {
            h[s] = h[s] * __expf(d * a[s]) + du * bc[DTR + s];
            y += h[s] * bc[DTR + DS + s];
        }
        ys[(size_t)t * DI + ch] = y;
    }
}

__global__ void k_ygate(const float* __restrict__ ys, const float* __restrict__ xcv,
                        const float* __restrict__ Dssm, const float* __restrict__ xz,
                        float* __restrict__ out)
{
    int i = blockIdx.x * blockDim.x + threadIdx.x;
    if (i >= L_SEQ * DI) return;
    int t = i / DI, ch = i % DI;
    float y = ys[i] + xcv[i] * Dssm[ch];
    float z = xz[(size_t)t * (2 * DI) + DI + ch];
    out[i] = y * siluf(z);
}

__global__ __launch_bounds__(256)
void k_layernorm(const float* __restrict__ src, const float* __restrict__ g,
                 const float* __restrict__ b, const float* __restrict__ res,
                 float* __restrict__ dst)
{
    __shared__ float red[256];
    const int row = blockIdx.x;
    const float* x = src + (size_t)row * DM;
    float s = 0.f, s2 = 0.f;
    for (int c = threadIdx.x; c < DM; c += 256) { float v = x[c]; s += v; s2 += v * v; }
    red[threadIdx.x] = s; __syncthreads();
    for (int o = 128; o > 0; o >>= 1) { if (threadIdx.x < o) red[threadIdx.x] += red[threadIdx.x + o]; __syncthreads(); }
    const float mean = red[0] / DM; __syncthreads();
    red[threadIdx.x] = s2; __syncthreads();
    for (int o = 128; o > 0; o >>= 1) { if (threadIdx.x < o) red[threadIdx.x] += red[threadIdx.x + o]; __syncthreads(); }
    const float var = red[0] / DM - mean * mean;
    const float inv = rsqrtf(var + 1e-5f);
    for (int c = threadIdx.x; c < DM; c += 256) {
        float v = (x[c] - mean) * inv * g[c] + b[c];
        if (res) v += res[(size_t)row * DM + c];
        dst[(size_t)row * DM + c] = v;
    }
}

__global__ void k_colmean(const float* __restrict__ m, float* __restrict__ mu)
{
    int c = blockIdx.x * blockDim.x + threadIdx.x;
    if (c >= DM) return;
    float s = 0.f;
    for (int t = 0; t < L_SEQ; ++t) s += m[(size_t)t * DM + c];
    mu[c] = s / (float)L_SEQ;
}

__global__ void k_gate1(const float* __restrict__ mu, const float* __restrict__ W1,
                        const float* __restrict__ b1, float* __restrict__ g1)
{
    int o = threadIdx.x;                          // 256
    float s = b1[o];
    for (int k = 0; k < DM; ++k) s += mu[k] * W1[k * 256 + o];
    g1[o] = geluf(s);
}

__global__ void k_gate2(const float* __restrict__ g1, const float* __restrict__ W2,
                        const float* __restrict__ b2, float* __restrict__ dw)
{
    __shared__ float lg[3];
    int t = threadIdx.x;
    if (t < 3) {
        float s = b2[t];
        for (int k = 0; k < 256; ++k) s += g1[k] * W2[k * 3 + t];
        lg[t] = s;
    }
    __syncthreads();
    if (t == 0) {
        float mx = fmaxf(lg[0], fmaxf(lg[1], lg[2]));
        float e0 = __expf(lg[0] - mx), e1 = __expf(lg[1] - mx), e2 = __expf(lg[2] - mx);
        float se = e0 + e1 + e2;
        dw[0] = e0 / se; dw[1] = e1 / se; dw[2] = e2 / se;
    }
}

__global__ void k_copy(const float* __restrict__ s, float* __restrict__ d, int n)
{ int i = blockIdx.x * blockDim.x + threadIdx.x; if (i < n) d[i] = s[i]; }

__global__ void k_zero(float* __restrict__ d, int n)
{ int i = blockIdx.x * blockDim.x + threadIdx.x; if (i < n) d[i] = 0.f; }

__global__ void k_add(float* __restrict__ d, const float* __restrict__ a, int n)
{ int i = blockIdx.x * blockDim.x + threadIdx.x; if (i < n) d[i] += a[i]; }

__global__ void k_concat2(const float* __restrict__ a, const float* __restrict__ b,
                          float* __restrict__ dst)
{
    int i = blockIdx.x * blockDim.x + threadIdx.x;
    if (i >= L_SEQ * 2 * DM) return;
    int r = i / (2 * DM), c = i % (2 * DM);
    dst[i] = (c < DM) ? a[(size_t)r * DM + c] : b[(size_t)r * DM + c - DM];
}

__global__ void k_update(float* __restrict__ out, float* __restrict__ state,
                         const float* __restrict__ t, const float* __restrict__ ns,
                         const float* __restrict__ dw, int d)
{
    int i = blockIdx.x * blockDim.x + threadIdx.x;
    if (i >= L_SEQ * DM) return;
    float w = dw[d];
    out[i]   += w * t[i];
    state[i] += w * ns[i];
}

__global__ void k_pool(const float* __restrict__ src, float* __restrict__ dst, int s, int Lp)
{
    int i = blockIdx.x * blockDim.x + threadIdx.x;
    if (i >= Lp * DM) return;
    int r = i / DM, c = i % DM;
    float acc = 0.f;
    for (int j = 0; j < s; ++j) acc += src[(size_t)(r * s + j) * DM + c];
    dst[i] = acc / (float)s;
}

__global__ void k_interp(const float* __restrict__ src, int Lp,
                         float* __restrict__ dst, int coloff)
{
    int i = blockIdx.x * blockDim.x + threadIdx.x;
    if (i >= L_SEQ * DM) return;
    int t = i / DM, c = i % DM;
    float pos = (t + 0.5f) * ((float)Lp / (float)L_SEQ) - 0.5f;
    pos = fminf(fmaxf(pos, 0.f), (float)(Lp - 1));
    int i0 = (int)floorf(pos);
    int i1 = min(i0 + 1, Lp - 1);
    float f = pos - (float)i0;
    dst[(size_t)t * (3 * DM) + coloff + c] =
        src[(size_t)i0 * DM + c] * (1.f - f) + src[(size_t)i1 * DM + c] * f;
}

// ---------------------------------------------------------------------------
// Orchestration
// ---------------------------------------------------------------------------
extern "C" void kernel_launch(void* const* d_in, const int* in_sizes, int n_in,
                              void* d_out, int out_size, void* d_ws, size_t ws_size,
                              hipStream_t stream)
{
    (void)in_sizes; (void)n_in; (void)out_size; (void)ws_size;
    const float* x       = (const float*)d_in[0];
    const float* W_in    = (const float*)d_in[1];
    const float* conv_w  = (const float*)d_in[2];
    const float* conv_b  = (const float*)d_in[3];
    const float* W_xproj = (const float*)d_in[4];
    const float* W_dt    = (const float*)d_in[5];
    const float* b_dt    = (const float*)d_in[6];
    const float* A_log   = (const float*)d_in[7];
    const float* D_ssm   = (const float*)d_in[8];
    const float* W_mout  = (const float*)d_in[9];
    const float* gW1     = (const float*)d_in[10];
    const float* gb1     = (const float*)d_in[11];
    const float* gW2     = (const float*)d_in[12];
    const float* gb2     = (const float*)d_in[13];
    const float* rlng    = (const float*)d_in[14];
    const float* rlnb    = (const float*)d_in[15];
    const float* rW1     = (const float*)d_in[16];
    const float* rb1     = (const float*)d_in[17];
    const float* rW2     = (const float*)d_in[18];
    const float* rb2     = (const float*)d_in[19];
    const float* memW    = (const float*)d_in[20];
    const float* memb    = (const float*)d_in[21];
    const float* haWp    = (const float*)d_in[22];
    const float* habp    = (const float*)d_in[23];
    const float* haWqkv  = (const float*)d_in[24];
    const float* habqkv  = (const float*)d_in[25];
    const float* haWo    = (const float*)d_in[26];
    const float* habo    = (const float*)d_in[27];
    const float* haWf    = (const float*)d_in[28];
    const float* habf    = (const float*)d_in[29];
    const float* fusW    = (const float*)d_in[30];
    const float* fusb    = (const float*)d_in[31];
    const float* fuslng  = (const float*)d_in[32];
    const float* fuslnb  = (const float*)d_in[33];

    float* ws = (float*)d_ws;
    size_t off = 0;
    auto alloc = [&](size_t n) { float* p = ws + off; off += n; return p; };
    float* xz    = alloc((size_t)L_SEQ * 2 * DI);
    float* xcv   = alloc((size_t)L_SEQ * DI);
    float* dBC   = alloc((size_t)L_SEQ * 64);
    float* delta = alloc((size_t)L_SEQ * DI);
    float* ys    = alloc((size_t)L_SEQ * DI);
    float* yg    = alloc((size_t)L_SEQ * DI);
    float* mbuf  = alloc((size_t)L_SEQ * DM);
    float* outb  = alloc((size_t)L_SEQ * DM);
    float* state = alloc((size_t)L_SEQ * DM);
    float* lnout = alloc((size_t)L_SEQ * DM);
    float* h1    = alloc((size_t)L_SEQ * 2 * DM);
    float* tb    = alloc((size_t)L_SEQ * DM);
    float* catb  = alloc((size_t)L_SEQ * 2 * DM);
    float* ns    = alloc((size_t)L_SEQ * DM);
    float* mu    = alloc(512);
    float* g1    = alloc(256);
    float* dw    = alloc(64);
    float* pbuf  = alloc((size_t)L_SEQ * DM);
    float* proj  = alloc((size_t)L_SEQ * DM);
    float* qkv   = alloc((size_t)L_SEQ * 3 * DM);
    float* ao    = alloc((size_t)L_SEQ * DM);
    float* mo    = alloc((size_t)L_SEQ * DM);
    float* acat  = alloc((size_t)L_SEQ * 3 * DM);
    float* ha    = alloc((size_t)L_SEQ * DM);
    float* cat2  = alloc((size_t)L_SEQ * 2 * DM);
    float* fus   = alloc((size_t)L_SEQ * DM);

    auto GEMM = [&](const float* A, int lda, const float* B, int ldb,
                    float* C, int ldc, const float* bias,
                    int M, int N, int K, int act) {
        dim3 grid(N / BN, M / BM);
        k_gemm<<<grid, GT, 0, stream>>>(A, lda, B, ldb, C, ldc, bias, M, N, K, act);
    };
    const int EW = 256;
    auto blocks = [&](int n) { return (n + EW - 1) / EW; };

    // ---- Mamba ----
    GEMM(x, DM, W_in, 2 * DI, xz, 2 * DI, nullptr, L_SEQ, 2 * DI, DM, 0);
    k_conv_silu<<<blocks(L_SEQ * DI), EW, 0, stream>>>(xz, conv_w, conv_b, xcv);
    GEMM(xcv, DI, W_xproj, 64, dBC, 64, nullptr, L_SEQ, 64, DI, 0);
    GEMM(dBC, 64, W_dt, DI, delta, DI, b_dt, L_SEQ, DI, DTR, 3);   // softplus(dt@W_dt+b)
    k_scan<<<8, 128, 0, stream>>>(delta, xcv, dBC, A_log, ys);
    k_ygate<<<blocks(L_SEQ * DI), EW, 0, stream>>>(ys, xcv, D_ssm, xz, yg);
    GEMM(yg, DI, W_mout, DM, mbuf, DM, nullptr, L_SEQ, DM, DI, 0);

    // ---- Recursive block ----
    k_colmean<<<blocks(DM), EW, 0, stream>>>(mbuf, mu);
    k_gate1<<<1, 256, 0, stream>>>(mu, gW1, gb1, g1);
    k_gate2<<<1, 32, 0, stream>>>(g1, gW2, gb2, dw);
    k_copy<<<blocks(L_SEQ * DM), EW, 0, stream>>>(mbuf, outb, L_SEQ * DM);
    k_zero<<<blocks(L_SEQ * DM), EW, 0, stream>>>(state, L_SEQ * DM);
    for (int d = 0; d < 3; ++d) {
        k_layernorm<<<L_SEQ, 256, 0, stream>>>(outb, rlng + d * DM, rlnb + d * DM, nullptr, lnout);
        GEMM(lnout, DM, rW1 + (size_t)d * DM * 2 * DM, 2 * DM, h1, 2 * DM,
             rb1 + d * 2 * DM, L_SEQ, 2 * DM, DM, 1);              // gelu
        GEMM(h1, 2 * DM, rW2 + (size_t)d * 2 * DM * DM, DM, tb, DM,
             rb2 + d * DM, L_SEQ, DM, 2 * DM, 0);
        k_concat2<<<blocks(L_SEQ * 2 * DM), EW, 0, stream>>>(tb, state, catb);
        GEMM(catb, 2 * DM, memW, DM, ns, DM, memb, L_SEQ, DM, 2 * DM, 0);
        k_update<<<blocks(L_SEQ * DM), EW, 0, stream>>>(outb, state, tb, ns, dw, d);
    }

    // ---- Hierarchical attention ----
    const int scales[3] = {1, 2, 4};
    for (int i = 0; i < 3; ++i) {
        const int s = scales[i], Lp = L_SEQ / s;
        const float* psrc = outb;
        if (s > 1) {
            k_pool<<<blocks(Lp * DM), EW, 0, stream>>>(outb, pbuf, s, Lp);
            psrc = pbuf;
        }
        GEMM(psrc, DM, haWp + (size_t)i * DM * DM, DM, proj, DM, habp + i * DM, Lp, DM, DM, 0);
        GEMM(proj, DM, haWqkv + (size_t)i * DM * 3 * DM, 3 * DM, qkv, 3 * DM,
             habqkv + i * 3 * DM, Lp, 3 * DM, DM, 0);
        dim3 ag(Lp / 16, NHEADS);
        k_attn<<<ag, 32, 0, stream>>>(qkv, Lp, ao);
        if (s == 1) {
            GEMM(ao, DM, haWo + (size_t)i * DM * DM, DM, acat, 3 * DM, habo + i * DM, Lp, DM, DM, 0);
        } else {
            GEMM(ao, DM, haWo + (size_t)i * DM * DM, DM, mo, DM, habo + i * DM, Lp, DM, DM, 0);
            k_interp<<<blocks(L_SEQ * DM), EW, 0, stream>>>(mo, Lp, acat, i * DM);
        }
    }
    GEMM(acat, 3 * DM, haWf, DM, ha, DM, habf, L_SEQ, DM, 3 * DM, 0);
    k_add<<<blocks(L_SEQ * DM), EW, 0, stream>>>(outb, ha, L_SEQ * DM);

    // ---- Fusion + outputs ----
    k_concat2<<<blocks(L_SEQ * 2 * DM), EW, 0, stream>>>(mbuf, outb, cat2);
    GEMM(cat2, 2 * DM, fusW, DM, fus, DM, fusb, L_SEQ, DM, 2 * DM, 0);
    k_layernorm<<<L_SEQ, 256, 0, stream>>>(fus, fuslng, fuslnb, x, (float*)d_out);
    k_copy<<<blocks(L_SEQ * DM), EW, 0, stream>>>(state, (float*)d_out + (size_t)L_SEQ * DM,
                                                  L_SEQ * DM);
}
